// LocallyConnected2d_11879879540889
// MI455X (gfx1250) — matrix operand order repaired and checked by reading, not compiled, not run on hardware
//
#include <hip/hip_runtime.h>
#include <hip/hip_bf16.h>

// LocallyConnected2d on MI455X (gfx1250, wave32):
// per spatial location s in [0,1024): Out[b,o] = sum_{ck<576} A[b,ck]*W[o,ck] + bias[o]
// A = 3x3 patches of x (zero padded), W = per-location weights.
// bf16 WMMA path: 151MB weight stream dominates (>= ~8us @ 23.3 TB/s); bf16
// tensor math keeps compute far under that bound.

typedef __attribute__((ext_vector_type(16))) __bf16 v16bf;
typedef __attribute__((ext_vector_type(8)))  __bf16 v8bf;
typedef __attribute__((ext_vector_type(8)))  float  v8f;

#define KTOT   576      // C*9
#define KPITCH 584      // padded LDS row stride (elements); 584 % 8 == 0 -> 16B aligned rows
#define SPAT   1024     // 32*32
#define OSTRIDE 589824  // weight o-stride in floats: 64*1024*9
#define CSTRIDE 9216    // weight c-stride in floats: 1024*9

__device__ __forceinline__ unsigned short f2bf(float f) {
    unsigned u = __builtin_bit_cast(unsigned, f);
    u += 0x7FFFu + ((u >> 16) & 1u);          // round-to-nearest-even
    return (unsigned short)(u >> 16);
}

__device__ __forceinline__ v16bf cat8(v8bf a, v8bf b) {
    return __builtin_shufflevector(a, b, 0, 1, 2, 3, 4, 5, 6, 7,
                                         8, 9, 10, 11, 12, 13, 14, 15);
}

__global__ __launch_bounds__(256)
void lc2d_wmma_bf16_kernel(const float* __restrict__ x,
                           const float* __restrict__ weight,
                           const float* __restrict__ bias,
                           float* __restrict__ out)
{
    extern __shared__ unsigned short lds[];          // A: [64][KPITCH], B: [64][KPITCH]
    unsigned short* __restrict__ ldsA = lds;
    unsigned short* __restrict__ ldsB = lds + 64 * KPITCH;

    const int s = blockIdx.x;                        // spatial index
    const int X = s >> 5;
    const int Y = s & 31;
    const int t = threadIdx.x;

    // ---------------- Phase 1: stage A (patches) and B (weights) as bf16 ----------------
    // 4096 (row, c) chunks of 9 elements per matrix; 16 chunks per thread each.
    #pragma unroll
    for (int i = 0; i < 16; ++i) {
        const int idx = t + (i << 8);                // 0..4095
        const int r  = idx >> 6;                     // batch b (for A) / out-ch o (for B)
        const int c  = idx & 63;                     // in-channel

        // A: 3x3 patch of x[b=r, c] centered at (X, Y), zero padded
        {
            const float* __restrict__ xb = x + (size_t)(((r << 6) + c) << 10);
            unsigned short* __restrict__ dst = ldsA + r * KPITCH + c * 9;
            #pragma unroll
            for (int kh = 0; kh < 3; ++kh) {
                const int h = X + kh - 1;
                #pragma unroll
                for (int kw = 0; kw < 3; ++kw) {
                    const int w = Y + kw - 1;
                    float v = 0.0f;
                    if ((unsigned)h < 32u && (unsigned)w < 32u)
                        v = xb[(h << 5) + w];
                    dst[kh * 3 + kw] = f2bf(v);
                }
            }
        }
        // B: weight[o=r, c, X, Y, 0..8] -> 9 contiguous floats
        {
            const float* __restrict__ wp =
                weight + (size_t)r * OSTRIDE + (size_t)c * CSTRIDE + (size_t)s * 9;
            unsigned short* __restrict__ dst = ldsB + r * KPITCH + c * 9;
            #pragma unroll
            for (int kk = 0; kk < 9; ++kk)
                dst[kk] = f2bf(wp[kk]);
        }
    }
    __syncthreads();

    // ---------------- Phase 2: 64x64x576 GEMM, V_WMMA_F32_16X16X32_BF16 ----------------
    const int wv   = t >> 5;            // wave id 0..7
    const int lane = t & 31;
    const int lo   = lane & 15;
    const int hi   = lane >> 4;

    const int m  = wv & 3;              // A row-tile (batch)
    const int n0 = (wv >> 2) << 1;      // first of two B col-tiles (out-ch)

    const int arow  = (m << 4) + lo;           // batch row this lane carries
    const int bcol0 = ((n0 + 0) << 4) + lo;    // out-ch column, tile 0
    const int bcol1 = ((n0 + 1) << 4) + lo;    // out-ch column, tile 1

    // A(16x32) lane map: elems 0..15 -> K = kb + hi*16 + 0..15
    const unsigned short* __restrict__ aBase  = ldsA + arow  * KPITCH + hi * 16;
    // B(32x16) lane map (contiguous, same as A): elems 0..15 -> K = kb + hi*16 + 0..15
    const unsigned short* __restrict__ b0Base = ldsB + bcol0 * KPITCH + hi * 16;
    const unsigned short* __restrict__ b1Base = ldsB + bcol1 * KPITCH + hi * 16;

    v8f acc0 = {};
    v8f acc1 = {};

    #pragma unroll 3
    for (int ks = 0; ks < 18; ++ks) {
        const int kb = ks << 5;
        const v8bf aLoV = *(const v8bf*)(aBase + kb);
        const v8bf aHiV = *(const v8bf*)(aBase + kb + 8);
        const v16bf afr = cat8(aLoV, aHiV);

        const v8bf b0a = *(const v8bf*)(b0Base + kb);
        const v8bf b0b = *(const v8bf*)(b0Base + kb + 8);
        const v16bf b0 = cat8(b0a, b0b);

        const v8bf b1a = *(const v8bf*)(b1Base + kb);
        const v8bf b1b = *(const v8bf*)(b1Base + kb + 8);
        const v16bf b1 = cat8(b1a, b1b);

        acc0 = __builtin_amdgcn_wmma_f32_16x16x32_bf16(
            false, afr, false, b0, (short)0, acc0, false, false);
        acc1 = __builtin_amdgcn_wmma_f32_16x16x32_bf16(
            false, afr, false, b1, (short)0, acc1, false, false);
    }

    // ---------------- Epilogue: bias + scattered fp32 stores ----------------
    // C/D layout: VGPR r -> M = m*16 + hi*8 + r, N = lane-constant column.
    const float bv0 = bias[(size_t)bcol0 * SPAT + s];
    const float bv1 = bias[(size_t)bcol1 * SPAT + s];

    const int mbase = (m << 4) + (hi << 3);
    #pragma unroll
    for (int r = 0; r < 8; ++r) {
        const size_t brow = (size_t)(mbase + r) << 16;          // b * 64 * 1024
        out[brow + (size_t)bcol0 * SPAT + s] = acc0[r] + bv0;
        out[brow + (size_t)bcol1 * SPAT + s] = acc1[r] + bv1;
    }
}

extern "C" void kernel_launch(void* const* d_in, const int* in_sizes, int n_in,
                              void* d_out, int out_size, void* d_ws, size_t ws_size,
                              hipStream_t stream) {
    (void)in_sizes; (void)n_in; (void)out_size; (void)d_ws; (void)ws_size;
    const float* x      = (const float*)d_in[0];
    const float* weight = (const float*)d_in[1];
    const float* bias   = (const float*)d_in[2];
    float* out          = (float*)d_out;

    const size_t shmem = (size_t)2 * 64 * KPITCH * sizeof(unsigned short); // ~149.5 KB
    dim3 grid(SPAT);     // one workgroup per spatial location
    dim3 block(256);     // 8 wave32
    lc2d_wmma_bf16_kernel<<<grid, block, shmem, stream>>>(x, weight, bias, out);
}
